// RNN_42451456753739
// MI455X (gfx1250) — matrix operand order repaired and checked by reading, not compiled, not run on hardware
//
#include <hip/hip_runtime.h>
#include <hip/hip_bf16.h>

// ---- problem constants -----------------------------------------------------
#define SEQ_L 200
#define BATCH 512
#define F_IN  6
#define HID   1024
#define BPAD  40        // padded K-stride (bf16 elems) of LDS B tile: 80B rows

typedef __attribute__((ext_vector_type(16))) __bf16 v16bf;
typedef __attribute__((ext_vector_type(8)))  __bf16 v8bf;
typedef __attribute__((ext_vector_type(8)))  float  v8f;
typedef __attribute__((ext_vector_type(4)))  int    v4i;

// Prefer the gfx1250 async global->LDS path when the toolchain exposes it.
#if defined(__has_builtin)
#if __has_builtin(__builtin_amdgcn_global_load_async_to_lds_b128) && \
    __has_builtin(__builtin_amdgcn_s_wait_asynccnt)
#define USE_ASYNC 1
#endif
#endif
#ifndef USE_ASYNC
#define USE_ASYNC 0
#endif

// ---------------------------------------------------------------------------
// prep: convert the three H x H fp32 weight matrices to bf16 and zero the
// initial hidden-state slots. Runs every launch (deterministic).
// grid: 4096 x 256 threads == HID*HID elements exactly.
// ---------------------------------------------------------------------------
__global__ void rnn_prep(const float* __restrict__ w0f,
                         const float* __restrict__ w1f,
                         const float* __restrict__ w2f,
                         __bf16* __restrict__ w0b,
                         __bf16* __restrict__ w1b,
                         __bf16* __restrict__ w2b,
                         __bf16* __restrict__ zero0,   // hs0 slot 0 (N*H)
                         __bf16* __restrict__ zero1) { // h1 slot 0  (N*H)
  const int i = blockIdx.x * blockDim.x + threadIdx.x;
  w0b[i] = (__bf16)w0f[i];
  w1b[i] = (__bf16)w1f[i];
  w2b[i] = (__bf16)w2f[i];
  if (i < BATCH * HID) {
    zero0[i] = (__bf16)0.0f;
    zero1[i] = (__bf16)0.0f;
  }
}

// ---------------------------------------------------------------------------
// Stage one 64(N) x 32(K) bf16 weight tile into LDS (padded rows of BPAD).
// 128 threads x 2 iters x 16B = 4KB. Async path uses GLOBAL_LOAD_ASYNC_TO_LDS
// (tracked by ASYNCcnt); fallback path is global_load_b128 + ds_store_b128.
// ---------------------------------------------------------------------------
__device__ __forceinline__ void stage_tile(const __bf16* __restrict__ W,
                                           int n0, int k0, int tid,
                                           __bf16* dst) {
#pragma unroll
  for (int it = 0; it < 2; ++it) {
    const int idx = tid + it * 128;
    const int row = idx >> 2;          // 0..63  (output column of W)
    const int c16 = (idx & 3) * 8;     // 0,8,16,24 (K sub-offset, 16B units)
    const __bf16* gsrc = W + (size_t)(n0 + row) * HID + (k0 + c16);
    __bf16* ldst = dst + row * BPAD + c16;
#if USE_ASYNC
    __builtin_amdgcn_global_load_async_to_lds_b128(
        (__attribute__((address_space(1))) v4i*)gsrc,
        (__attribute__((address_space(3))) v4i*)ldst, 0, 0);
#else
    *(v8bf*)ldst = *(const v8bf*)gsrc;
#endif
  }
}

// ---------------------------------------------------------------------------
// One K=HID accumulation pass of D += A[16x1024] * W^T using
// v_wmma_f32_16x16x32_bf16. Per-wave tile: 16 rows (M) x 64 cols (N).
// B tiles are staged through a double-buffered LDS tile shared by all 4
// waves (cuts B vector-memory traffic 4x); A is loaded directly from global.
//
// A layout (16-bit A 16x32, ISA 7.12.2): lane l<16 -> row M=l, bf16[0..7]=K
// kb..kb+7, bf16[8..15]=K kb+16..kb+23 with kb=0; lane l+16 -> same row,
// kb=8. B layout (16-bit B 32x16): col N=lane&15, same K order as A
// (bf16[0..7]=K kb..kb+7, bf16[8..15]=K kb+16..kb+23).
// ---------------------------------------------------------------------------
__device__ __forceinline__ void gemm_pass(const __bf16* __restrict__ A,
                                          const __bf16* __restrict__ W,
                                          int m0, int n0, int tid,
                                          __bf16* sB, v8f acc[4]) {
  const int lane = tid & 31;
  const int r  = m0 + (lane & 15);          // A row
  const int kb = (lane >> 4) * 8;           // A K sub-offset
  const int cl = lane & 15;                 // B column within 16-tile
  const int kr = (lane >> 4) * 8;           // B K sub-offset

  stage_tile(W, n0, 0, tid, sB);
#if USE_ASYNC
  __builtin_amdgcn_s_wait_asynccnt(0);
#endif
  __syncthreads();

  for (int kc = 0; kc < HID / 32; ++kc) {
    const int cur = kc & 1;
    // Prefetch the next K-chunk into the other LDS buffer.
    if (kc + 1 < HID / 32)
      stage_tile(W, n0, (kc + 1) * 32, tid, sB + (1 - cur) * (64 * BPAD));

    union { v16bf v; v8bf h[2]; } ua;
    const __bf16* ap = A + (size_t)r * HID + (kc * 32 + kb);
    ua.h[0] = *(const v8bf*)(ap);
    ua.h[1] = *(const v8bf*)(ap + 16);

    const __bf16* base = sB + cur * (64 * BPAD);
#pragma unroll
    for (int j = 0; j < 4; ++j) {
      union { v16bf v; v8bf h[2]; } ub;
      const v8bf* bp = (const v8bf*)(base + (cl + j * 16) * BPAD + kr);
      ub.h[0] = bp[0];
      ub.h[1] = bp[2];
      acc[j] = __builtin_amdgcn_wmma_f32_16x16x32_bf16(
          /*neg_a=*/false, ua.v, /*neg_b=*/false, ub.v,
          /*c_mod=*/(short)0, acc[j], /*reuse_a=*/false, /*reuse_b=*/false);
    }
#if USE_ASYNC
    __builtin_amdgcn_s_wait_asynccnt(0);
#endif
    __syncthreads();
  }
}

// ---------------------------------------------------------------------------
// One RNN timestep for one layer:
//   Hout = tanh( A1 @ W1^T  [+ A2 @ W2^T]  [+ seq_t @ Wih^T]  + b1 + b2 )
// Layer 0: A1 = hs0[t-1], W1 = W_hh0(bf16), inproj from seq_t/Wih0 (K=6).
// Layer 1: A1 = hs0[t],   W1 = W_ih1(bf16), A2 = h1[t-1], W2 = W_hh1(bf16).
// Block: 128 threads = 4 waves; block tile 64(M) x 64(N); grid (8, 16).
// ---------------------------------------------------------------------------
__global__ __launch_bounds__(128)
void rnn_step(const __bf16* __restrict__ A1, const __bf16* __restrict__ W1,
              const __bf16* __restrict__ A2, const __bf16* __restrict__ W2,
              const float*  __restrict__ seq_t,  // [BATCH, F_IN] or null
              const float*  __restrict__ Wih,    // [HID, F_IN]   or null
              const float*  __restrict__ b1, const float* __restrict__ b2,
              __bf16* __restrict__ Hout) {
  __shared__ __bf16 sB[2 * 64 * BPAD];  // double-buffered 64x32 B tile (10KB)

  const int tid  = threadIdx.x;
  const int lane = tid & 31;
  const int wave = tid >> 5;
  const int m0 = blockIdx.x * 64 + wave * 16;
  const int n0 = blockIdx.y * 64;

  v8f acc[4];
#pragma unroll
  for (int j = 0; j < 4; ++j) acc[j] = {};

  gemm_pass(A1, W1, m0, n0, tid, sB, acc);
  if (A2) gemm_pass(A2, W2, m0, n0, tid, sB, acc);

  // C/D layout: lane half selects M-block (rows rbase..rbase+7), lane&15 = N.
  const int rbase = m0 + ((lane >> 4) * 8);
  const int cl = lane & 15;

  float sx[8][6];
  if (seq_t) {
#pragma unroll
    for (int v = 0; v < 8; ++v)
#pragma unroll
      for (int f = 0; f < F_IN; ++f)
        sx[v][f] = seq_t[(size_t)(rbase + v) * F_IN + f];
  }

#pragma unroll
  for (int j = 0; j < 4; ++j) {
    const int c = n0 + j * 16 + cl;
    const float bb = b1[c] + b2[c];
    float wv[6];
    if (seq_t) {
#pragma unroll
      for (int f = 0; f < F_IN; ++f) wv[f] = Wih[(size_t)c * F_IN + f];
    }
#pragma unroll
    for (int v = 0; v < 8; ++v) {
      float x = acc[j][v] + bb;
      if (seq_t) {
#pragma unroll
        for (int f = 0; f < F_IN; ++f) x += sx[v][f] * wv[f];
      }
      Hout[(size_t)(rbase + v) * HID + c] = (__bf16)tanhf(x);
    }
  }
}

// ---------------------------------------------------------------------------
// Output head: out[n] = dot(h1_last[n,:], W_out) + b_out   (fp32 result)
// ---------------------------------------------------------------------------
__global__ void rnn_head(const __bf16* __restrict__ h1,
                         const float* __restrict__ Wout,
                         const float* __restrict__ bout,
                         float* __restrict__ out) {
  const int n = blockIdx.x * blockDim.x + threadIdx.x;
  if (n < BATCH) {
    float s = 0.0f;
    for (int h = 0; h < HID; ++h)
      s += (float)h1[(size_t)n * HID + h] * Wout[h];
    out[n] = s + bout[0];
  }
}

// ---------------------------------------------------------------------------
extern "C" void kernel_launch(void* const* d_in, const int* in_sizes, int n_in,
                              void* d_out, int out_size, void* d_ws, size_t ws_size,
                              hipStream_t stream) {
  const float* seq  = (const float*)d_in[0];   // [L, N, F_IN]
  const float* Wih0 = (const float*)d_in[1];   // [H, F_IN]
  const float* Whh0 = (const float*)d_in[2];   // [H, H]
  const float* bih0 = (const float*)d_in[3];
  const float* bhh0 = (const float*)d_in[4];
  const float* Wih1 = (const float*)d_in[5];   // [H, H]
  const float* Whh1 = (const float*)d_in[6];   // [H, H]
  const float* bih1 = (const float*)d_in[7];
  const float* bhh1 = (const float*)d_in[8];
  const float* Wout = (const float*)d_in[9];   // [1, H]
  const float* bout = (const float*)d_in[10];  // [1]

  const size_t NH = (size_t)BATCH * HID;
  // Workspace layout (bf16, all regions 16B+ aligned):
  //   hs0 : (L+1) slots of [N,H]; slot 0 = zero initial state, slot t+1 = hs0[t]
  //   h1  : 2 ping-pong slots of [N,H]
  //   wh0 / wi1 / wh1 : bf16 copies of the three H x H weights
  __bf16* hs0 = (__bf16*)d_ws;
  __bf16* h1  = hs0 + (size_t)(SEQ_L + 1) * NH;
  __bf16* wh0 = h1  + 2 * NH;
  __bf16* wi1 = wh0 + (size_t)HID * HID;
  __bf16* wh1 = wi1 + (size_t)HID * HID;

  rnn_prep<<<(HID * HID) / 256, 256, 0, stream>>>(
      Whh0, Wih1, Whh1, wh0, wi1, wh1, hs0, h1);

  const dim3 grid(BATCH / 64, HID / 64);  // 8 x 16 workgroups
  for (int t = 0; t < SEQ_L; ++t) {
    // Layer 0: hs0[t] = tanh(hs0[t-1] @ Whh0^T + seq[t] @ Wih0^T + b)
    rnn_step<<<grid, 128, 0, stream>>>(
        hs0 + (size_t)t * NH, wh0, nullptr, nullptr,
        seq + (size_t)t * BATCH * F_IN, Wih0, bih0, bhh0,
        hs0 + (size_t)(t + 1) * NH);
    // Layer 1: h1[t] = tanh(hs0[t] @ Wih1^T + h1[t-1] @ Whh1^T + b)
    rnn_step<<<grid, 128, 0, stream>>>(
        hs0 + (size_t)(t + 1) * NH, wi1,
        h1 + (size_t)(t & 1) * NH, wh1,
        nullptr, nullptr, bih1, bhh1,
        h1 + (size_t)((t + 1) & 1) * NH);
  }
  // After t = L-1 the final h1 lives in slot (L & 1) == 0.
  rnn_head<<<(BATCH + 255) / 256, 256, 0, stream>>>(h1, Wout, bout, (float*)d_out);
}
